// MixtureRNN_69054484185307
// MI455X (gfx1250) — matrix operand order repaired and checked by reading, not compiled, not run on hardware
//
#include <hip/hip_runtime.h>
#include <stdint.h>

// Problem constants (from reference): B=64, T=512, F=32, H=512, O=680
#define BB   64
#define TT   512
#define FF   32
#define HH   512
#define OO   680
#define K1   (FF + HH)        // 544  = 17 * 32
#define G1   (4 * HH)         // 2048
#define OP   688              // layer-2 per-gate width padded to 43*16
#define G2P  (4 * OP)         // 2752 padded gate rows
#define K2   1216             // [c1(512) | h2(680) + pad(24)] = 38 * 32
#define X2W  1216             // row width of layer-2 input buffer
#define H2OFF HH              // h2 starts at col 512 inside x2 buffer

typedef __attribute__((ext_vector_type(16))) __bf16        v16bf;
typedef __attribute__((ext_vector_type(8)))  float         v8f;
typedef __attribute__((ext_vector_type(8)))  unsigned int  v8u;
typedef __attribute__((ext_vector_type(4)))  unsigned int  v4u;
typedef __attribute__((ext_vector_type(8)))  int           v8i;
typedef __attribute__((ext_vector_type(4)))  int           v4i;

#if defined(__has_builtin)
#  if __has_builtin(__builtin_amdgcn_tensor_load_to_lds)
#    define HAVE_TDM 1
#  else
#    define HAVE_TDM 0
#  endif
#else
#  define HAVE_TDM 0
#endif
#if __has_include(<hip/amd_detail/amd_gfx1250_TDM.h>)
#  define TDM_6ARG 1
#else
#  define TDM_6ARG 0
#endif

__device__ __forceinline__ unsigned short f2bf(float f) {
    unsigned int u = __builtin_bit_cast(unsigned int, f);
    u += 0x7FFFu + ((u >> 16) & 1u);        // round-to-nearest-even
    return (unsigned short)(u >> 16);
}
__device__ __forceinline__ float sigm(float x) { return 1.0f / (1.0f + __expf(-x)); }
__device__ __forceinline__ float tanh_(float x) {
    float ax = fabsf(x);
    float e  = __expf(-2.0f * ax);
    float r  = (1.0f - e) / (1.0f + e);
    return x < 0.0f ? -r : r;
}

// A-operand: 16-bit A 16x32 layout. Lane (l,hi): halves j -> k = hi*16 + j
// (matches B) -> two contiguous 16B loads.
__device__ __forceinline__ v16bf ld_a(const unsigned short* p0, const unsigned short* p1) {
    v8u t;
    ((uint4*)&t)[0] = *(const uint4*)p0;
    ((uint4*)&t)[1] = *(const uint4*)p1;
    return __builtin_bit_cast(v16bf, t);
}
// B-operand: 16-bit B 32x16. Lane (n=l, hi): halves j -> k = hi*16 + j.
// 32 contiguous bytes per lane (works for global or LDS pointers).
__device__ __forceinline__ v16bf ld_b(const unsigned short* p) {
    v8u t;
    ((uint4*)&t)[0] = ((const uint4*)p)[0];
    ((uint4*)&t)[1] = ((const uint4*)p)[1];
    return __builtin_bit_cast(v16bf, t);
}

#define WMMA_BF16(A, B, C) \
    __builtin_amdgcn_wmma_f32_16x16x32_bf16(false, (A), false, (B), (short)0, (C), false, false)

// ---------------------------------------------------------------- TDM staging
// One 3D TDM op stages a [4 gate][16 row][32 k] bf16 panel (4KB) into LDS.
// D# packing per CDNA5 ISA (data_size=1 -> 2 bytes).
#if HAVE_TDM
__device__ __forceinline__ void tdm_b_panel(unsigned int lds_byte_off,
                                            const unsigned short* gbase,
                                            unsigned long long rowStride,    // elems
                                            unsigned long long gateStride) { // elems
    unsigned long long ga = (unsigned long long)(uintptr_t)gbase;
    v4u g0;
    g0[0] = 1u;                                                   // count=1, is_restore=0
    g0[1] = lds_byte_off;                                         // lds_addr
    g0[2] = (unsigned int)ga;                                     // global_addr[31:0]
    g0[3] = ((unsigned int)(ga >> 32) & 0x01ffffffu) | (2u << 30); // addr[56:32] | type=2
    v8i g1;
    g1[0] = (int)(1u << 16);                                      // wg_mask=0, data_size=2B
    g1[1] = (int)(32u << 16);                                     // tensor_dim0 = 32
    g1[2] = (int)(16u << 16);                                     // dim0 hi=0 | tensor_dim1 = 16
    g1[3] = (int)(32u << 16);                                     // dim1 hi=0 | tile_dim0 = 32
    g1[4] = (int)(16u | (4u << 16));                              // tile_dim1=16 | tile_dim2=4
    g1[5] = (int)(unsigned int)rowStride;                         // tensor_dim0_stride lo32
    g1[6] = (int)(((unsigned int)(rowStride >> 32) & 0xffffu) |
                  ((unsigned int)(gateStride & 0xffffu) << 16));  // d0s hi | d1s lo16
    g1[7] = (int)(unsigned int)(gateStride >> 16);                // tensor_dim1_stride[47:16]
    v4i g2; g2[0] = 4; g2[1] = 0; g2[2] = 0; g2[3] = 0;           // tensor_dim2=4
    v4i g3 = {0, 0, 0, 0};
#if TDM_6ARG
    v8i g4 = {0, 0, 0, 0, 0, 0, 0, 0};
    __builtin_amdgcn_tensor_load_to_lds(g0, g1, g2, g3, g4, 0);
#else
    __builtin_amdgcn_tensor_load_to_lds(g0, g1, g2, g3, 0);
#endif
}
#endif

// LDS byte offset of a generic shared pointer (aperture: low 32 bits = LDS addr)
__device__ __forceinline__ unsigned int lds_off(const void* p) {
    return (unsigned int)(uintptr_t)p;
}

// ---------------------------------------------------------------- prep kernels

__global__ void k_cvt_data(const float* __restrict__ src, unsigned short* __restrict__ dst, int n) {
    int i = blockIdx.x * blockDim.x + threadIdx.x;
    if (i < n) dst[i] = f2bf(src[i]);
}

__global__ void k_build_w1(const float* __restrict__ Wih1, const float* __restrict__ Whh1,
                           unsigned short* __restrict__ W1) {
    int i = blockIdx.x * blockDim.x + threadIdx.x;
    if (i >= G1 * K1) return;
    int nrow = i / K1, k = i - nrow * K1;
    float v = (k < FF) ? Wih1[nrow * FF + k] : Whh1[(size_t)nrow * HH + (k - FF)];
    W1[i] = f2bf(v);
}

__global__ void k_build_w2(const float* __restrict__ Wih2, const float* __restrict__ Whh2,
                           unsigned short* __restrict__ W2) {
    int i = blockIdx.x * blockDim.x + threadIdx.x;
    if (i >= G2P * K2) return;
    int pr = i / K2, k = i - pr * K2;
    int g = pr / OP, r = pr - g * OP;
    float v = 0.0f;
    if (r < OO && k < HH + OO) {
        int nrow = g * OO + r;
        v = (k < HH) ? Wih2[(size_t)nrow * HH + k] : Whh2[(size_t)nrow * OO + (k - HH)];
    }
    W2[i] = f2bf(v);
}

__global__ void k_misc(const float* __restrict__ bih1, const float* __restrict__ bhh1,
                       const float* __restrict__ bih2, const float* __restrict__ bhh2,
                       float* __restrict__ bias1, float* __restrict__ bias2,
                       unsigned short* __restrict__ h1a, unsigned short* __restrict__ h1b,
                       unsigned short* __restrict__ x2a, unsigned short* __restrict__ x2b,
                       float* __restrict__ c1, float* __restrict__ c2) {
    int i = blockIdx.x * blockDim.x + threadIdx.x;
    if (i < G1) bias1[i] = bih1[i] + bhh1[i];
    if (i < G2P) {
        int g = i / OP, r = i - g * OP;
        bias2[i] = (r < OO) ? (bih2[g * OO + r] + bhh2[g * OO + r]) : 0.0f;
    }
    if (i < BB * HH) { h1a[i] = 0; h1b[i] = 0; c1[i] = 0.0f; }
    if (i < BB * X2W) { x2a[i] = 0; x2b[i] = 0; }
    if (i < BB * OO) c2[i] = 0.0f;
}

// ---------------------------------------------------------------- step kernels
// Block = 4 waves (mt=0..3 batch tiles) sharing one 16-wide output-column tile.
// B panels (identical across the 4 waves) are staged once per block into LDS,
// double-buffered: TDM (wave 0) when available, else cooperative gate-per-wave copy.

// Stage chunk kc of the weight panel into bsh[buf]
#define STAGE_PANEL(bsh, buf, wbase, kc, ROWSTRIDE, GATESTRIDE)                          \
    do {                                                                                 \
        if (HAVE_TDM) {                                                                  \
            if (wave == 0)                                                               \
                TDM_CALL(lds_off(&(bsh)[buf][0][0][0]), (wbase) + (size_t)(kc) * 32,     \
                         (unsigned long long)(ROWSTRIDE), (unsigned long long)(GATESTRIDE)); \
        } else {                                                                         \
            const unsigned short* s_ = (wbase) + (size_t)wave * (GATESTRIDE) +           \
                                       (size_t)l * (ROWSTRIDE) + (size_t)(kc) * 32 + hi * 16; \
            uint4 q0_ = ((const uint4*)s_)[0];                                           \
            uint4 q1_ = ((const uint4*)s_)[1];                                           \
            uint4* d_ = (uint4*)&(bsh)[buf][wave][l][hi * 16];                           \
            d_[0] = q0_; d_[1] = q1_;                                                    \
        }                                                                                \
    } while (0)

#if HAVE_TDM
#  define TDM_CALL(off, base, rs, gs) tdm_b_panel((off), (base), (rs), (gs))
#  define STAGE_WAIT()                                        \
      do {                                                    \
          if (wave == 0) __builtin_amdgcn_s_wait_tensorcnt(0); \
          __syncthreads();                                    \
      } while (0)
#else
#  define TDM_CALL(off, base, rs, gs) ((void)0)
#  define STAGE_WAIT() __syncthreads()
#endif

// Layer 1: gates(64x2048) = [x_t | h1](64x544) @ W1^T
__global__ void lstm1_step(const unsigned short* __restrict__ dataBf,
                           const unsigned short* __restrict__ W1,
                           const float* __restrict__ bias1,
                           const unsigned short* __restrict__ h1r,
                           unsigned short* __restrict__ h1w,
                           float* __restrict__ c1,
                           unsigned short* __restrict__ x2w,
                           int t) {
    __shared__ __align__(16) unsigned short bsh[2][4][16][32];  // [buf][gate][row][k]

    const int lane = threadIdx.x & 31;
    const int wave = threadIdx.x >> 5;   // = mt, 0..3
    const int ht   = blockIdx.x;         // 0..31 hidden tile
    const int l    = lane & 15;
    const int hi   = lane >> 4;
    const int m    = wave * 16 + l;      // batch row this lane loads (A operand)

    v8f acc0 = {0,0,0,0,0,0,0,0}, acc1 = {0,0,0,0,0,0,0,0};
    v8f acc2 = {0,0,0,0,0,0,0,0}, acc3 = {0,0,0,0,0,0,0,0};

    const unsigned short* wbase = W1 + (size_t)(ht * 16) * K1;

    STAGE_PANEL(bsh, 0, wbase, 0, K1, (size_t)HH * K1);
    STAGE_WAIT();

    for (int kc = 0; kc < K1 / 32; ++kc) {       // 17 chunks; chunk 0 = x_t, rest = h1
        const int cur = kc & 1;
        if (kc + 1 < K1 / 32)
            STAGE_PANEL(bsh, cur ^ 1, wbase, kc + 1, K1, (size_t)HH * K1);

        const unsigned short* p0 = (kc == 0)
            ? (dataBf + (size_t)m * (TT * FF) + (size_t)t * FF)
            : (h1r + (size_t)m * HH + (kc - 1) * 32);
        v16bf a = ld_a(p0 + hi * 16, p0 + hi * 16 + 8);
        acc0 = WMMA_BF16(a, ld_b(&bsh[cur][0][l][hi * 16]), acc0);
        acc1 = WMMA_BF16(a, ld_b(&bsh[cur][1][l][hi * 16]), acc1);
        acc2 = WMMA_BF16(a, ld_b(&bsh[cur][2][l][hi * 16]), acc2);
        acc3 = WMMA_BF16(a, ld_b(&bsh[cur][3][l][hi * 16]), acc3);

        if (kc + 1 < K1 / 32) STAGE_WAIT();
    }

    const int n = ht * 16 + l;                   // hidden column of this lane
    const float bi = bias1[0 * HH + n];
    const float bf = bias1[1 * HH + n];
    const float bg = bias1[2 * HH + n];
    const float bo = bias1[3 * HH + n];
    #pragma unroll
    for (int r = 0; r < 8; ++r) {
        const int mo = wave * 16 + hi * 8 + r;   // C/D layout: lanes16-31 -> M=8+r
        float ig = sigm (acc0[r] + bi);
        float fg = sigm (acc1[r] + bf);
        float gg = tanh_(acc2[r] + bg);
        float og = sigm (acc3[r] + bo);
        float cold = c1[(size_t)mo * HH + n];
        float cn = fg * cold + ig * gg;
        float hn = og * tanh_(cn);
        c1[(size_t)mo * HH + n]   = cn;
        h1w[(size_t)mo * HH + n]  = f2bf(hn);
        x2w[(size_t)mo * X2W + n] = f2bf(cn);    // c1 feeds layer 2
    }
}

// Layer 2: gates(64x2752pad) = [c1 | h2](64x1216) @ W2^T
__global__ void lstm2_step(const unsigned short* __restrict__ W2,
                           const float* __restrict__ bias2,
                           const unsigned short* __restrict__ x2r,
                           unsigned short* __restrict__ x2w,
                           float* __restrict__ c2,
                           float* __restrict__ out,
                           int t) {
    __shared__ __align__(16) unsigned short bsh[2][4][16][32];

    const int lane = threadIdx.x & 31;
    const int wave = threadIdx.x >> 5;   // = mt, 0..3
    const int ot   = blockIdx.x;         // 0..42
    const int l    = lane & 15;
    const int hi   = lane >> 4;
    const int m    = wave * 16 + l;

    v8f acc0 = {0,0,0,0,0,0,0,0}, acc1 = {0,0,0,0,0,0,0,0};
    v8f acc2 = {0,0,0,0,0,0,0,0}, acc3 = {0,0,0,0,0,0,0,0};

    const unsigned short* abase = x2r + (size_t)m * X2W;
    const unsigned short* wbase = W2 + (size_t)(ot * 16) * K2;

    STAGE_PANEL(bsh, 0, wbase, 0, K2, (size_t)OP * K2);
    STAGE_WAIT();

    for (int kc = 0; kc < K2 / 32; ++kc) {       // 38 chunks
        const int cur = kc & 1;
        if (kc + 1 < K2 / 32)
            STAGE_PANEL(bsh, cur ^ 1, wbase, kc + 1, K2, (size_t)OP * K2);

        const unsigned short* p0 = abase + kc * 32;
        v16bf a = ld_a(p0 + hi * 16, p0 + hi * 16 + 8);
        acc0 = WMMA_BF16(a, ld_b(&bsh[cur][0][l][hi * 16]), acc0);
        acc1 = WMMA_BF16(a, ld_b(&bsh[cur][1][l][hi * 16]), acc1);
        acc2 = WMMA_BF16(a, ld_b(&bsh[cur][2][l][hi * 16]), acc2);
        acc3 = WMMA_BF16(a, ld_b(&bsh[cur][3][l][hi * 16]), acc3);

        if (kc + 1 < K2 / 32) STAGE_WAIT();
    }

    const int n = ot * 16 + l;                   // padded output col 0..687
    const bool valid = (n < OO);
    const float bi = bias2[0 * OP + n];
    const float bf = bias2[1 * OP + n];
    const float bg = bias2[2 * OP + n];
    const float bo = bias2[3 * OP + n];
    #pragma unroll
    for (int r = 0; r < 8; ++r) {
        const int mo = wave * 16 + hi * 8 + r;
        float ig = sigm (acc0[r] + bi);
        float fg = sigm (acc1[r] + bf);
        float gg = tanh_(acc2[r] + bg);
        float og = sigm (acc3[r] + bo);
        float cold = valid ? c2[(size_t)mo * OO + n] : 0.0f;
        float cn = fg * cold + ig * gg;
        float hn = og * tanh_(cn);
        if (valid) {
            c2[(size_t)mo * OO + n] = cn;
            out[(size_t)mo * TT * OO + (size_t)t * OO + n] = cn;   // output is c2
            x2w[(size_t)mo * X2W + H2OFF + n] = f2bf(hn);          // h2 for next step
        }
    }
}

// ---------------------------------------------------------------- host

extern "C" void kernel_launch(void* const* d_in, const int* in_sizes, int n_in,
                              void* d_out, int out_size, void* d_ws, size_t ws_size,
                              hipStream_t stream) {
    const float* data = (const float*)d_in[0];
    const float* Wih1 = (const float*)d_in[1];
    const float* Whh1 = (const float*)d_in[2];
    const float* bih1 = (const float*)d_in[3];
    const float* bhh1 = (const float*)d_in[4];
    const float* Wih2 = (const float*)d_in[5];
    const float* Whh2 = (const float*)d_in[6];
    const float* bih2 = (const float*)d_in[7];
    const float* bhh2 = (const float*)d_in[8];
    float* out = (float*)d_out;

    uint8_t* p = (uint8_t*)d_ws;
    auto carve = [&](size_t bytes) -> void* {
        void* r = (void*)p;
        p += (bytes + 255) & ~(size_t)255;
        return r;
    };
    unsigned short* dataBf = (unsigned short*)carve((size_t)BB * TT * FF * 2);   // 2 MB
    unsigned short* W1     = (unsigned short*)carve((size_t)G1 * K1 * 2);        // 2.2 MB
    unsigned short* W2     = (unsigned short*)carve((size_t)G2P * K2 * 2);       // 6.7 MB
    float*          bias1  = (float*)carve((size_t)G1 * 4);
    float*          bias2  = (float*)carve((size_t)G2P * 4);
    unsigned short* h1a    = (unsigned short*)carve((size_t)BB * HH * 2);
    unsigned short* h1b    = (unsigned short*)carve((size_t)BB * HH * 2);
    unsigned short* x2a    = (unsigned short*)carve((size_t)BB * X2W * 2);
    unsigned short* x2b    = (unsigned short*)carve((size_t)BB * X2W * 2);
    float*          c1     = (float*)carve((size_t)BB * HH * 4);
    float*          c2     = (float*)carve((size_t)BB * OO * 4);

    const int nData = BB * TT * FF;
    k_cvt_data<<<(nData + 255) / 256, 256, 0, stream>>>(data, dataBf, nData);
    k_build_w1<<<(G1 * K1 + 255) / 256, 256, 0, stream>>>(Wih1, Whh1, W1);
    k_build_w2<<<(G2P * K2 + 255) / 256, 256, 0, stream>>>(Wih2, Whh2, W2);
    k_misc<<<(BB * X2W + 255) / 256, 256, 0, stream>>>(bih1, bhh1, bih2, bhh2,
                                                       bias1, bias2, h1a, h1b, x2a, x2b, c1, c2);

    unsigned short* h1buf[2] = { h1a, h1b };
    unsigned short* x2buf[2] = { x2a, x2b };
    for (int t = 0; t < TT; ++t) {
        const int par = t & 1;
        lstm1_step<<<HH / 16, 128, 0, stream>>>(dataBf, W1, bias1,
                                                h1buf[par], h1buf[par ^ 1], c1,
                                                x2buf[par], t);
        lstm2_step<<<OP / 16, 128, 0, stream>>>(W2, bias2,
                                                x2buf[par], x2buf[par ^ 1],
                                                c2, out, t);
    }
}
